// BayesianSkipgram_68049461837850
// MI455X (gfx1250) — matrix operand order repaired and checked by reading, not compiled, not run on hardware
//
#include <hip/hip_runtime.h>
#include <hip/hip_bf16.h>
#include <math.h>

typedef __attribute__((ext_vector_type(16))) __bf16 v16bf;
typedef __attribute__((ext_vector_type(8)))  float  v8f;

#define VOCAB 50257
#define DIM   128
#define BATCH 2048
#define CTX   10

// ---- workspace layout (float offsets) ----
#define OFF_HSUM  0          // B*D floats
#define OFF_Z     262144     // B*D floats
#define OFF_KL    524288     // B floats
#define OFF_CTXL  526336     // B*C floats
#define OFF_PARTM 546816     // 128*32*16 floats
#define OFF_PARTS 612352     // 128*32*16 floats
#define OFF_ZBF   677888     // B*D ushorts (131072 floats)
#define OFF_WGBF  808960     // N*D ushorts (3216448 floats)

#define NTILES 3142          // ceil(50257/16)
#define NSPLIT 32
#define TPS    99            // ceil(NTILES/NSPLIT)

#define NEG_BIG (-3.0e38f)

union BFfrag { v16bf v; uint4 q[2]; };

__device__ __forceinline__ unsigned short f2bf(float f) {
    unsigned int u = __float_as_uint(f);
    u += 0x7FFFu + ((u >> 16) & 1u);         // round-to-nearest-even
    return (unsigned short)(u >> 16);
}

// branchless (m,s) merge; finite sentinels only (NEG_BIG, never -inf)
__device__ __forceinline__ void merge_ms(float& m, float& s, float om, float os) {
    float nm = fmaxf(m, om);
    s = s * __expf(m - nm) + os * __expf(om - nm);
    m = nm;
}

// ---------------- K1: W_gen fp32 -> bf16 ----------------
__global__ void k_wgen_bf16(const float* __restrict__ wg, unsigned short* __restrict__ out) {
    int i = blockIdx.x * 256 + threadIdx.x;
    if (i < VOCAB * DIM) out[i] = f2bf(wg[i]);
}

// ---------------- K2: inference net, h_sum --------------
__global__ void k_hsum(const int* __restrict__ xb, const int* __restrict__ cw,
                       const float* __restrict__ emb, const float* __restrict__ Waff,
                       const float* __restrict__ baff, float* __restrict__ hsum) {
    extern __shared__ float sm[];
    float* sW   = sm;                 // 128*257
    float* sCen = sW + 128 * 257;     // 128
    float* sCtx = sCen + 128;         // 10*128
    const int b = blockIdx.x, d = threadIdx.x;

    for (int i = d; i < 128 * 256; i += 128) {
        int r = i >> 8, c = i & 255;
        sW[r * 257 + c] = Waff[i];
    }
    int xc = xb[b];
    sCen[d] = emb[xc * DIM + d];
    for (int c = 0; c < CTX; ++c)
        sCtx[c * 128 + d] = emb[cw[b * CTX + c] * DIM + d];
    __syncthreads();

    const float* wr = sW + d * 257;
    float accc = baff[d];
    #pragma unroll 4
    for (int e = 0; e < 128; ++e) accc += sCen[e] * wr[e];
    float hs = 0.f;
    for (int c = 0; c < CTX; ++c) {
        const float* cx = sCtx + c * 128;
        float t = accc;
        #pragma unroll 4
        for (int e = 0; e < 128; ++e) t += cx[e] * wr[128 + e];
        hs += fmaxf(t, 0.f);
    }
    hsum[b * DIM + d] = hs;
}

// ---------------- K3: mu/sigma/z/KL ---------------------
__global__ void k_z_kl(const int* __restrict__ xb, const float* __restrict__ eps,
                       const float* __restrict__ hsum,
                       const float* __restrict__ Wmu, const float* __restrict__ bmu,
                       const float* __restrict__ Wsig, const float* __restrict__ bsig,
                       const float* __restrict__ gsig,
                       float* __restrict__ z, unsigned short* __restrict__ zbf,
                       float* __restrict__ kl) {
    extern __shared__ float sm[];
    float* sWm = sm;                 // 128*129
    float* sWs = sWm + 128 * 129;    // 128*129
    float* sH  = sWs + 128 * 129;    // 128
    float* sR  = sH + 128;           // 4
    const int b = blockIdx.x, d = threadIdx.x;

    for (int i = d; i < 128 * 128; i += 128) {
        int r = i >> 7, c = i & 127;
        sWm[r * 129 + c] = Wmu[i];
        sWs[r * 129 + c] = Wsig[i];
    }
    sH[d] = hsum[b * DIM + d];
    __syncthreads();

    const float* wm = sWm + d * 129;
    const float* ws = sWs + d * 129;
    float mu = bmu[d], sg = bsig[d];
    #pragma unroll 4
    for (int e = 0; e < 128; ++e) { float h = sH[e]; mu += h * wm[e]; sg += h * ws[e]; }
    float isig = fmaxf(sg, 0.f) + log1pf(__expf(-fabsf(sg)));       // softplus
    float zz = mu + eps[b * DIM + d] * isig;
    z[b * DIM + d] = zz;
    zbf[b * DIM + d] = f2bf(zz);

    float gp = gsig[xb[b] * DIM + d];
    float gsg = fmaxf(gp, 0.f) + log1pf(__expf(-fabsf(gp)));
    float dmu = mu - gsg;
    float kd = logf(gsg / isig) + (isig * isig + dmu * dmu) / (2.f * gsg * gsg) - 0.5f;

    for (int off = 16; off; off >>= 1) kd += __shfl_xor(kd, off, 32);
    if ((d & 31) == 0) sR[d >> 5] = kd;
    __syncthreads();
    if (d == 0) kl[b] = sR[0] + sR[1] + sR[2] + sR[3];
}

// ---------------- K4: exact context logits (fp32) -------
__global__ void k_ctx_logits(const int* __restrict__ cw, const float* __restrict__ z,
                             const float* __restrict__ Wgen, const float* __restrict__ bgen,
                             float* __restrict__ ctxl) {
    int idx  = blockIdx.x * 4 + (threadIdx.x >> 5);   // (b,c) pair per wave
    int lane = threadIdx.x & 31;
    int b = idx / CTX, c = idx % CTX;
    int w = cw[b * CTX + c];
    const float* zr = z + b * DIM;
    const float* wr = Wgen + (long)w * DIM;
    float acc = 0.f;
    #pragma unroll
    for (int k = 0; k < 4; ++k) acc += zr[lane + 32 * k] * wr[lane + 32 * k];
    for (int off = 16; off; off >>= 1) acc += __shfl_xor(acc, off, 32);
    if (lane == 0) ctxl[idx] = acc + bgen[w];
}

// ---------------- K5: WMMA bf16 GEMM + fixed-base logsumexp -
__device__ __forceinline__ v8f tile_logits(int t, int lm, int half,
        const BFfrag a[4], const unsigned short* __restrict__ wgbf,
        const float* __restrict__ bgen, float& bg) {
    int n  = t * 16 + lm;
    int nc = n < VOCAB ? n : VOCAB - 1;
    const unsigned short* wr = wgbf + (long)nc * DIM + half * 8; // lanes16-31 -> K 8-15 / 24-31
    BFfrag bf[4];
    #pragma unroll
    for (int kc = 0; kc < 4; ++kc) {
        bf[kc].q[0] = *(const uint4*)(wr + kc * 32);
        bf[kc].q[1] = *(const uint4*)(wr + kc * 32 + 16);
    }
    bg = (n < VOCAB) ? bgen[nc] : NEG_BIG;   // tail lanes poisoned -> exp()=0, max no-op
    v8f c = {};
    #pragma unroll
    for (int kc = 0; kc < 4; ++kc)
        c = __builtin_amdgcn_wmma_f32_16x16x32_bf16(false, a[kc].v, false, bf[kc].v,
                                                    (short)0, c, false, false);
    return c;
}

// grid = 128 batch-tiles * 32 N-splits; 128 threads (4 waves)
__global__ void k_lse(const unsigned short* __restrict__ zbf,
                      const unsigned short* __restrict__ wgbf,
                      const float* __restrict__ bgen,
                      float* __restrict__ partM, float* __restrict__ partS) {
    const int btile = blockIdx.x >> 5;
    const int split = blockIdx.x & 31;
    const int tid = threadIdx.x, wid = tid >> 5, lane = tid & 31;
    const int lm = lane & 15, half = lane >> 4;

    // A fragments: 16 rows of z (bf16), K=128 in 4 chunks, reused across all n-tiles
    BFfrag a[4];
    const unsigned short* zrow = zbf + (btile * 16 + lm) * DIM;
    #pragma unroll
    for (int kc = 0; kc < 4; ++kc) {
        a[kc].q[0] = *(const uint4*)(zrow + kc * 32 + half * 8);        // K 0-7 / 8-15
        a[kc].q[1] = *(const uint4*)(zrow + kc * 32 + 16 + half * 8);   // K 16-23 / 24-31
    }

    float M[8], tm[8], s[8];

    const int t0 = split * TPS + wid;
    int t1 = split * TPS + TPS; if (t1 > NTILES) t1 = NTILES;

    // peel first tile (always fully valid: only tile 3141 has tail, never first)
    {
        float bg; v8f c = tile_logits(t0, lm, half, a, wgbf, bgen, bg);
        #pragma unroll
        for (int r = 0; r < 8; ++r) { float v = c[r] + bg; M[r] = v; tm[r] = v; s[r] = 1.f; }
    }

    // software-pipelined by one stage: update uses the PREVIOUS tile's accumulator,
    // so the exp/max VALU work fills the current WMMA chain's hazard shadow.
    int t = t0 + 4;
    if (t < t1) {
        float bgp; v8f cp = tile_logits(t, lm, half, a, wgbf, bgen, bgp);
        int it = 0;
        for (t += 4; t < t1; t += 4) {
            int tn = t + 4;
            if (tn < t1) __builtin_prefetch(wgbf + (long)(tn * 16 + lm) * DIM, 0, 1);
            float bgc; v8f cc = tile_logits(t, lm, half, a, wgbf, bgen, bgc);
            #pragma unroll
            for (int r = 0; r < 8; ++r) {
                float v = cp[r] + bgp;
                tm[r] = fmaxf(tm[r], v);
                s[r] += __expf(v - M[r]);    // 1 exp/element, 8 independent chains
            }
            if ((++it & 7) == 0) {           // uniform branch: periodic rebase
                #pragma unroll
                for (int r = 0; r < 8; ++r) { s[r] *= __expf(M[r] - tm[r]); M[r] = tm[r]; }
            }
            cp = cc; bgp = bgc;
        }
        #pragma unroll
        for (int r = 0; r < 8; ++r) {        // drain last tile
            float v = cp[r] + bgp;
            tm[r] = fmaxf(tm[r], v);
            s[r] += __expf(v - M[r]);
        }
    }
    #pragma unroll
    for (int r = 0; r < 8; ++r) { s[r] *= __expf(M[r] - tm[r]); M[r] = tm[r]; }

    // branchless reduce over 16 columns per half-wave
    for (int off = 1; off < 16; off <<= 1) {
        #pragma unroll
        for (int r = 0; r < 8; ++r) {
            float om = __shfl_xor(M[r], off, 32);
            float os = __shfl_xor(s[r], off, 32);
            merge_ms(M[r], s[r], om, os);
        }
    }
    __shared__ float sM[4][16], sS[4][16];
    if (lm == 0) {  // lanes 0 (rows 0-7) and 16 (rows 8-15)
        #pragma unroll
        for (int r = 0; r < 8; ++r) { sM[wid][r + half * 8] = M[r]; sS[wid][r + half * 8] = s[r]; }
    }
    __syncthreads();
    if (tid < 16) {
        float m = sM[0][tid], ss = sS[0][tid];
        for (int w = 1; w < 4; ++w) merge_ms(m, ss, sM[w][tid], sS[w][tid]);
        int o = (btile * NSPLIT + split) * 16 + tid;
        partM[o] = m; partS[o] = ss;
    }
}

// ---------------- K6: deterministic finalize ------------
__global__ void k_finalize(const float* __restrict__ partM, const float* __restrict__ partS,
                           const float* __restrict__ ctxl, const float* __restrict__ kl,
                           float* __restrict__ out) {
    __shared__ float sR[8];
    float local = 0.f;
    for (int b = threadIdx.x; b < BATCH; b += 256) {
        int bt = b >> 4, row = b & 15;
        float m = NEG_BIG, s = 0.f;
        for (int sp = 0; sp < NSPLIT; ++sp) {
            int o = (bt * NSPLIT + sp) * 16 + row;
            merge_ms(m, s, partM[o], partS[o]);
        }
        float lse = m + logf(s);
        float recon = -((float)CTX) * lse;
        for (int c = 0; c < CTX; ++c) recon += ctxl[b * CTX + c];
        local += kl[b] - recon;
    }
    for (int off = 16; off; off >>= 1) local += __shfl_xor(local, off, 32);
    if ((threadIdx.x & 31) == 0) sR[threadIdx.x >> 5] = local;
    __syncthreads();
    if (threadIdx.x == 0) {
        float t = 0.f;
        for (int w = 0; w < 8; ++w) t += sR[w];
        out[0] = t / (float)BATCH;
    }
}

extern "C" void kernel_launch(void* const* d_in, const int* in_sizes, int n_in,
                              void* d_out, int out_size, void* d_ws, size_t ws_size,
                              hipStream_t stream) {
    (void)in_sizes; (void)n_in; (void)out_size; (void)ws_size;
    const int*   xb   = (const int*)  d_in[0];
    const int*   cw   = (const int*)  d_in[1];
    const float* eps  = (const float*)d_in[2];
    const float* emb  = (const float*)d_in[3];
    const float* Waff = (const float*)d_in[4];
    const float* baff = (const float*)d_in[5];
    const float* Wmu  = (const float*)d_in[6];
    const float* bmu  = (const float*)d_in[7];
    const float* Wsig = (const float*)d_in[8];
    const float* bsig = (const float*)d_in[9];
    const float* gsig = (const float*)d_in[10];
    const float* Wgen = (const float*)d_in[11];
    const float* bgen = (const float*)d_in[12];
    float* out = (float*)d_out;

    float* ws = (float*)d_ws;
    float* hsum  = ws + OFF_HSUM;
    float* z     = ws + OFF_Z;
    float* kl    = ws + OFF_KL;
    float* ctxl  = ws + OFF_CTXL;
    float* partM = ws + OFF_PARTM;
    float* partS = ws + OFF_PARTS;
    unsigned short* zbf  = (unsigned short*)(ws + OFF_ZBF);
    unsigned short* wgbf = (unsigned short*)(ws + OFF_WGBF);

    k_wgen_bf16<<<(VOCAB * DIM + 255) / 256, 256, 0, stream>>>(Wgen, wgbf);

    size_t sm1 = (128 * 257 + 128 + CTX * 128) * sizeof(float);
    k_hsum<<<BATCH, 128, sm1, stream>>>(xb, cw, emb, Waff, baff, hsum);

    size_t sm2 = (2 * 128 * 129 + 128 + 4) * sizeof(float);
    k_z_kl<<<BATCH, 128, sm2, stream>>>(xb, eps, hsum, Wmu, bmu, Wsig, bsig, gsig,
                                        z, zbf, kl);

    k_ctx_logits<<<BATCH * CTX / 4, 128, 0, stream>>>(cw, z, Wgen, bgen, ctxl);

    k_lse<<<(BATCH / 16) * NSPLIT, 128, 0, stream>>>(zbf, wgbf, bgen, partM, partS);

    k_finalize<<<1, 256, 0, stream>>>(partM, partS, ctxl, kl, out);
}
